// dynamicLSTM_29343216566479
// MI455X (gfx1250) — matrix operand order repaired and checked by reading, not compiled, not run on hardware
//
#include <hip/hip_runtime.h>
#include <math.h>

// ---------------------------------------------------------------------------
// Types for CDNA5 WMMA
// ---------------------------------------------------------------------------
typedef __bf16 bf16;
typedef __attribute__((ext_vector_type(16))) __bf16 v16bf;
typedef __attribute__((ext_vector_type(8)))  __bf16 v8bf;
typedef __attribute__((ext_vector_type(8)))  float  v8f;

__device__ __forceinline__ bf16 f2bf(float f) {
  union { float f; unsigned u; } v; v.f = f;
  unsigned r = (v.u + 0x7FFFu + ((v.u >> 16) & 1u)) >> 16;
  unsigned short s = (unsigned short)r;
  return __builtin_bit_cast(bf16, s);
}

__device__ __forceinline__ float sigm(float x) { return 1.0f / (1.0f + __expf(-x)); }

// Problem constants
#define NB   256          // NUM_STATES * b = 32*8
#define B_   8
#define K_   32
#define HC   128          // H_DIM
#define CIN  256          // 2*H_DIM (x || h)
#define SP   1024         // 32*32
#define OC4  512          // 4*H_DIM

// ---------------------------------------------------------------------------
// Kernel 0a: stage W_conv [512][256][3][3] fp32 -> sw[tap][oc][ic] bf16
// ---------------------------------------------------------------------------
__global__ __launch_bounds__(256) void k_stage_w(const float* __restrict__ Wc,
                                                 bf16* __restrict__ sw) {
  const int oc = blockIdx.x;            // 0..511
  for (int e = threadIdx.x; e < CIN * 9; e += 256) {
    int ci = e / 9, tap = e % 9;
    sw[((size_t)tap * OC4 + oc) * CIN + ci] = f2bf(Wc[((size_t)oc * CIN + ci) * 9 + tap]);
  }
}

// ---------------------------------------------------------------------------
// Kernel 0b: stage (x || h0) NCHW fp32 -> sxh[n][y][x][ci] NHWC bf16
// ---------------------------------------------------------------------------
__global__ __launch_bounds__(256) void k_stage_x(const float* __restrict__ inp,
                                                 const float* __restrict__ h0,
                                                 bf16* __restrict__ sxh) {
  const int n = blockIdx.x;             // 0..255  (n = k*b + bi)
  const int y = blockIdx.y;             // 0..31
  const int bi = n & 7;
  const int t = threadIdx.x;
  __shared__ float tile[CIN * 32];      // [ci][x]  32 KB

  for (int pass = 0; pass < 32; ++pass) {
    int ci = pass * 8 + (t >> 5);
    int x  = t & 31;
    float v;
    if (ci < HC) v = inp[((size_t)(bi * HC + ci)) * SP + y * 32 + x];
    else         v = h0 [((size_t)(n  * HC + (ci - HC))) * SP + y * 32 + x];
    tile[ci * 32 + x] = v;
  }
  __syncthreads();
  // write: consecutive threads write consecutive ci ranges at one x
  const int x = t >> 3;
  const int cib = (t & 7) * 32;
  bf16* dst = sxh + (((size_t)n * SP + y * 32 + x) << 8) + cib;
  for (int j = 0; j < 32; ++j) dst[j] = f2bf(tile[(cib + j) * 32 + x]);
}

// ---------------------------------------------------------------------------
// Kernel 1: fused gate conv (bf16 WMMA implicit GEMM) + ConvLSTM pointwise
// grid = (n=256, hcTile=8, sBlock=4), block = 256 (8 waves)
// Each wave: one image row (y = y0 + wave), two 16-col tiles (N=32), M=16 oc
// per gate, K = 256 ic x 9 taps. A operands reused across the two N tiles.
// dynamic LDS: act tile 10*34*256 bf16 (174080 B) + double-buffered per-tap
// weights (2 x 32768 B) = 239616 B (fits 320 KB WGP budget).
// ---------------------------------------------------------------------------
#define ACT_BYTES 174080
#define WBUF_ELEMS 16384          // 32768 B per buffer
#define LDS_BYTES (ACT_BYTES + 2 * 32768)

__global__ __launch_bounds__(256) void k_gates(const bf16* __restrict__ sxh,
                                               const bf16* __restrict__ sw,
                                               const float* __restrict__ bconv,
                                               const float* __restrict__ c0,
                                               float* __restrict__ h1ws,
                                               float* __restrict__ c1ws) {
  extern __shared__ char smem[];
  bf16* act  = (bf16*)smem;                 // [rr(10)][cc(34)][ci(256)]
  bf16* wbuf = (bf16*)(smem + ACT_BYTES);   // 2 x [gate(4)][hc(16)][ic(256)]

  const int n   = blockIdx.x;
  const int hc0 = blockIdx.y * 16;
  const int y0  = blockIdx.z * 8;

  const int t    = threadIdx.x;
  const int wv   = t >> 5;                  // wave = row within block (0..7)
  const int lane = t & 31;
  const int mrow = lane & 15;               // M (A rows) / N (B cols in tile)
  const int hi   = lane >> 4;               // lane half

  // ---- cooperative act tile load (10 rows x 34 cols, zero-padded halo) ----
  for (int cch = t; cch < 10 * 34 * 32; cch += 256) {
    int pos = cch >> 5;                     // (rr,cc)
    int sub = cch & 31;                     // 16B sub-chunk
    int rr = pos / 34, cc = pos % 34;
    int y = y0 - 1 + rr, x = cc - 1;
    uint4 v = {0u, 0u, 0u, 0u};
    if (y >= 0 && y < 32 && x >= 0 && x < 32)
      v = *(const uint4*)(sxh + (((size_t)n * SP + y * 32 + x) << 8) + (sub << 3));
    *(uint4*)((char*)act + (size_t)cch * 16) = v;
  }

  // ---- initial weight slice (tap 0) into buffer 0 ----
  for (int cch = t; cch < 2048; cch += 256) {
    int gidx = cch >> 9;
    int rem  = cch & 511;
    int hcr  = rem >> 5;
    int icc  = (rem & 31) << 3;
    uint4 v = *(const uint4*)(sw + ((size_t)(gidx * HC + hc0 + hcr)) * CIN + icc);
    *(uint4*)((char*)wbuf + (size_t)cch * 16) = v;
  }
  __syncthreads();

  v8f acc0[4], acc1[4];                     // [gate] for N-tile 0 / 1
  #pragma unroll
  for (int g = 0; g < 4; ++g) { acc0[g] = (v8f)(0.0f); acc1[g] = (v8f)(0.0f); }

  for (int tap = 0; tap < 9; ++tap) {
    bf16* wcur = wbuf + (tap & 1) * WBUF_ELEMS;
    bf16* wnxt = wbuf + ((tap + 1) & 1) * WBUF_ELEMS;

    // ---- prefetch next tap's weight slice into registers (latency hidden
    //      under the 64 WMMAs below) ----
    uint4 wreg[8];
    if (tap < 8) {
      #pragma unroll
      for (int q = 0; q < 8; ++q) {
        int cch = t + q * 256;
        int gidx = cch >> 9;
        int rem  = cch & 511;
        int hcr  = rem >> 5;
        int icc  = (rem & 31) << 3;
        wreg[q] = *(const uint4*)(sw +
            ((size_t)((tap + 1) * OC4 + gidx * HC + hc0 + hcr)) * CIN + icc);
      }
    }

    const int dy = tap / 3, dx = tap % 3;
    const bf16* b0   = act + (((wv + dy) * 34 + (mrow + dx)) << 8) + (hi << 4);
    const bf16* b1   = b0 + (16 << 8);      // +16 columns
    const bf16* wrow = wcur + (mrow << 8) + (hi << 4);

    #pragma unroll
    for (int icb = 0; icb < 8; ++icb) {
      // issue ALL operand loads first -> single dscnt wait, 8 WMMAs back-to-back
      v16bf B0 = *(const v16bf*)(b0 + icb * 32);
      v16bf B1 = *(const v16bf*)(b1 + icb * 32);
      v16bf Am[4];
      #pragma unroll
      for (int g = 0; g < 4; ++g) {
        const bf16* wp = wrow + g * 4096 + icb * 32;
        v8bf a0 = *(const v8bf*)(wp);
        v8bf a1 = *(const v8bf*)(wp + 8);
        Am[g] = __builtin_shufflevector(a0, a1, 0,1,2,3,4,5,6,7,8,9,10,11,12,13,14,15);
      }
      #pragma unroll
      for (int g = 0; g < 4; ++g) {
        acc0[g] = __builtin_amdgcn_wmma_f32_16x16x32_bf16(
            false, Am[g], false, B0, (short)0, acc0[g], false, false);
        acc1[g] = __builtin_amdgcn_wmma_f32_16x16x32_bf16(
            false, Am[g], false, B1, (short)0, acc1[g], false, false);
      }
    }

    // ---- commit prefetched weights to the alternate buffer ----
    if (tap < 8) {
      #pragma unroll
      for (int q = 0; q < 8; ++q)
        *(uint4*)((char*)wnxt + (size_t)(t + q * 256) * 16) = wreg[q];
    }
    __syncthreads();
  }

  // ---- fused LSTM pointwise epilogue (two N tiles) ----
  const int y = y0 + wv;
  #pragma unroll
  for (int T = 0; T < 2; ++T) {
    const v8f* acc = T ? acc1 : acc0;
    const int x = T * 16 + mrow;
    const int sp = y * 32 + x;
    #pragma unroll
    for (int r = 0; r < 8; ++r) {
      int hc = hc0 + (hi << 3) + r;
      float fv = acc[0][r] + bconv[hc];
      float iv = acc[1][r] + bconv[HC + hc];
      float ov = acc[2][r] + bconv[2 * HC + hc];
      float gv = acc[3][r] + bconv[3 * HC + hc];
      size_t cidx = ((size_t)n * HC + hc) * SP + sp;
      float c0v = c0[cidx];
      float lrg = gv > 0.0f ? gv : 0.01f * gv;          // leaky_relu
      float c1v = sigm(fv) * c0v + sigm(iv) * lrg;
      float h1v = sigm(ov) * tanhf(c1v);
      c1ws[cidx] = c1v;
      h1ws[cidx] = h1v;
    }
  }
}

// ---------------------------------------------------------------------------
// Kernel 2: observation conv (oc=1) + FC dot -> lp[n]   (tiny, VALU)
// ---------------------------------------------------------------------------
__global__ __launch_bounds__(256) void k_obs(const float* __restrict__ inp,
                                             const float* __restrict__ h1ws,
                                             const float* __restrict__ Wobs,
                                             const float* __restrict__ bobs,
                                             const float* __restrict__ Wfc,
                                             const float* __restrict__ bfc,
                                             float* __restrict__ lp) {
  const int n = blockIdx.x;
  const int bi = n & 7;
  const int t = threadIdx.x;
  float acc = 0.0f;
  for (int j = 0; j < 4; ++j) {
    int pos = t + j * 256;
    int y = pos >> 5, x = pos & 31;
    float s = bobs[0];
    for (int tap = 0; tap < 9; ++tap) {
      int yy = y + tap / 3 - 1, xx = x + tap % 3 - 1;
      if (yy < 0 || yy > 31 || xx < 0 || xx > 31) continue;
      int sp = yy * 32 + xx;
      for (int ci = 0; ci < HC; ++ci)
        s += inp[((size_t)(bi * HC + ci)) * SP + sp] * Wobs[ci * 9 + tap];
      for (int ci = 0; ci < HC; ++ci)
        s += h1ws[((size_t)(n * HC + ci)) * SP + sp] * Wobs[(HC + ci) * 9 + tap];
    }
    acc += s * Wfc[pos];
  }
  __shared__ float red[256];
  red[t] = acc; __syncthreads();
  for (int s2 = 128; s2 > 0; s2 >>= 1) {
    if (t < s2) red[t] += red[t + s2];
    __syncthreads();
  }
  if (t == 0) lp[n] = red[0] + bfc[0];
}

// ---------------------------------------------------------------------------
// Kernel 3: particle weight update + Gumbel-max categorical resampling
// single block of 256 threads; deterministic PCG hash for Gumbel noise
// ---------------------------------------------------------------------------
__device__ __forceinline__ unsigned pcg(unsigned x) {
  x = x * 747796405u + 2891336453u;
  unsigned w = ((x >> ((x >> 28u) + 4u)) ^ x) * 277803737u;
  return (w >> 22u) ^ w;
}

__global__ __launch_bounds__(256) void k_particle(const float* __restrict__ lp,
                                                  const float* __restrict__ p0,
                                                  int* __restrict__ flat,
                                                  float* __restrict__ pnws) {
  __shared__ float p1[NB];
  __shared__ float lse[B_];
  __shared__ float logits[NB];   // index k*8+bi
  __shared__ float pns[NB];
  const int t = threadIdx.x;     // n = k*b + bi
  const int bi = t & 7;

  p1[t] = lp[t] * p0[t];
  __syncthreads();
  if (t < B_) {
    float m = -1e30f;
    for (int k = 0; k < K_; ++k) m = fmaxf(m, p1[k * 8 + t]);
    float s = 0.0f;
    for (int k = 0; k < K_; ++k) s += __expf(p1[k * 8 + t] - m);
    lse[t] = m + __logf(s);
  }
  __syncthreads();
  float pv = p1[t] - lse[bi];
  p1[t] = pv;
  float resamp = 0.5f * __expf(pv) + 0.5f / (float)K_;
  logits[t] = __logf(resamp);
  __syncthreads();

  // categorical: argmax over j of logits[j*8+bi] + Gumbel(hash)
  float best = -1e30f; int bidx = 0;
  for (int j = 0; j < K_; ++j) {
    unsigned h = pcg((unsigned)(t * K_ + j) ^ 0x9E3779B9u);
    float u = ((float)h + 0.5f) * 2.3283064365386963e-10f;
    float gum = -__logf(-__logf(u));
    float v = logits[j * 8 + bi] + gum;
    if (v > best) { best = v; bidx = j; }
  }
  const int fl = bidx * 8 + bi;
  flat[t] = fl;
  __syncthreads();

  float pe = __expf(p1[fl]);
  float pn = __logf(pe / (0.5f * pe + 0.5f / (float)K_));
  pns[t] = pn;
  __syncthreads();
  if (t < B_) {
    float m = -1e30f;
    for (int k = 0; k < K_; ++k) m = fmaxf(m, pns[k * 8 + t]);
    float s = 0.0f;
    for (int k = 0; k < K_; ++k) s += __expf(pns[k * 8 + t] - m);
    lse[t] = m + __logf(s);
  }
  __syncthreads();
  pnws[t] = pns[t] - lse[bi];
}

// ---------------------------------------------------------------------------
// Kernel 4: gather resampled h1/c1 rows into d_out, emit pn
// grid = (n=256, chunk=32), block = 256
// ---------------------------------------------------------------------------
#define ROWF 131072                 // 128*1024 floats per row
#define OH1  0
#define OC1  33554432
#define OPN  67108864
#define OHID 67109120

__global__ __launch_bounds__(256) void k_gather(const float* __restrict__ h1ws,
                                                const float* __restrict__ c1ws,
                                                const int* __restrict__ flat,
                                                const float* __restrict__ pnws,
                                                float* __restrict__ out) {
  const int n = blockIdx.x, z = blockIdx.y;
  const int src = flat[n];
  const float4* hs = (const float4*)(h1ws + (size_t)src * ROWF) + (size_t)z * 1024;
  const float4* cs = (const float4*)(c1ws + (size_t)src * ROWF) + (size_t)z * 1024;
  float4* hd = (float4*)(out + OH1 + (size_t)n * ROWF) + (size_t)z * 1024;
  float4* cd = (float4*)(out + OC1 + (size_t)n * ROWF) + (size_t)z * 1024;
  for (int i = threadIdx.x; i < 1024; i += 256) { hd[i] = hs[i]; cd[i] = cs[i]; }
  if (z == 0 && threadIdx.x == 0) out[OPN + n] = pnws[n];
}

// ---------------------------------------------------------------------------
// Kernel 5: weighted mean state (reference's (b,K,-1) reinterpretation)
// hidden[bi][e] = sum_k h1n[bi*32+k][e] * exp(pn[bi*32+k])
// ---------------------------------------------------------------------------
__global__ __launch_bounds__(256) void k_hidden(const float* __restrict__ out_h1,
                                                const float* __restrict__ pnws,
                                                float* __restrict__ out) {
  const int bi = blockIdx.x;        // 0..7
  const int ch = blockIdx.y;        // 0..127 chunks of 1024
  __shared__ float w[K_];
  if (threadIdx.x < K_) w[threadIdx.x] = __expf(pnws[bi * K_ + threadIdx.x]);
  __syncthreads();
  for (int e = threadIdx.x; e < 1024; e += 256) {
    size_t idx = (size_t)ch * 1024 + e;
    float s = 0.0f;
    for (int k = 0; k < K_; ++k)
      s += out_h1[OH1 + (size_t)(bi * K_ + k) * ROWF + idx] * w[k];
    out[OHID + (size_t)bi * ROWF + idx] = s;
  }
}

// ---------------------------------------------------------------------------
// Launcher
// ---------------------------------------------------------------------------
extern "C" void kernel_launch(void* const* d_in, const int* in_sizes, int n_in,
                              void* d_out, int out_size, void* d_ws, size_t ws_size,
                              hipStream_t stream) {
  const float* inp   = (const float*)d_in[0];
  const float* h0    = (const float*)d_in[1];
  const float* c0    = (const float*)d_in[2];
  const float* p0    = (const float*)d_in[3];
  const float* Wconv = (const float*)d_in[4];
  const float* bconv = (const float*)d_in[5];
  const float* Wobs  = (const float*)d_in[6];
  const float* bobs  = (const float*)d_in[7];
  const float* Wfc   = (const float*)d_in[8];
  const float* bfc   = (const float*)d_in[9];
  float* out = (float*)d_out;

  // workspace carve-up
  char* ws = (char*)d_ws;
  bf16*  sxh  = (bf16*)ws;                                  // 134,217,728 B
  bf16*  sw   = (bf16*)(ws + 134217728);                    //   2,359,296 B
  float* h1ws = (float*)(ws + 134217728 + 2359296);         // 134,217,728 B
  float* c1ws = h1ws + (size_t)NB * ROWF;                   // 134,217,728 B
  float* lp   = c1ws + (size_t)NB * ROWF;                   // 1 KB
  int*   flat = (int*)(lp + NB);                            // 1 KB
  float* pnws = (float*)(flat + NB);                        // 1 KB

  k_stage_w<<<dim3(OC4), dim3(256), 0, stream>>>(Wconv, sw);
  k_stage_x<<<dim3(NB, 32), dim3(256), 0, stream>>>(inp, h0, sxh);
  k_gates<<<dim3(NB, 8, 4), dim3(256), LDS_BYTES, stream>>>(sxh, sw, bconv, c0, h1ws, c1ws);
  k_obs<<<dim3(NB), dim3(256), 0, stream>>>(inp, h1ws, Wobs, bobs, Wfc, bfc, lp);
  k_particle<<<dim3(1), dim3(256), 0, stream>>>(lp, p0, flat, pnws);
  k_gather<<<dim3(NB, 32), dim3(256), 0, stream>>>(h1ws, c1ws, flat, pnws, out);
  k_hidden<<<dim3(B_, 128), dim3(256), 0, stream>>>(out, pnws, out);
}
